// GaussianEmbedding_55508157333878
// MI455X (gfx1250) — matrix operand order repaired and checked by reading, not compiled, not run on hardware
//
#include <hip/hip_runtime.h>

#define NN2048 2048
#define BATCH 8
#define NCH 64

typedef __attribute__((ext_vector_type(8)))  float  v8f;
typedef __attribute__((ext_vector_type(8)))  __bf16 v8bf;
typedef __attribute__((ext_vector_type(16))) __bf16 v16bf;

// ---------------------------------------------------------------------------
// f32 -> bf16 hi/lo split (round-to-nearest-even on both parts).
// a ~= hi + lo ; dropping lo*lo in the product keeps ~2^-16 relative error.
// ---------------------------------------------------------------------------
__device__ __forceinline__ void split_bf16(float x, __bf16& hi, __bf16& lo) {
    unsigned u  = __float_as_uint(x);
    unsigned hb = (u + 0x7FFFu + ((u >> 16) & 1u)) & 0xFFFF0000u;
    float    hf = __uint_as_float(hb);
    float    l  = x - hf;
    unsigned ul = __float_as_uint(l);
    unsigned lb = (ul + 0x7FFFu + ((ul >> 16) & 1u)) >> 16;
    hi = __builtin_bit_cast(__bf16, (unsigned short)(hb >> 16));
    lo = __builtin_bit_cast(__bf16, (unsigned short)lb);
}

// A-fragment: lane = kHi*16+m holds
// K = 16*kHi .. 16*kHi+15 of row m (contiguous, matches B).
__device__ __forceinline__ v16bf lds_frag_a(const __bf16* p) {
    v8bf a = *(const v8bf*)p;          // 16B ds_load_b128
    v8bf b = *(const v8bf*)(p + 8);    // 16B ds_load_b128
    return __builtin_shufflevector(a, b, 0,1,2,3,4,5,6,7,8,9,10,11,12,13,14,15);
}
// B-fragment (B staged transposed in LDS): lane = kHi*16+n holds
// K = 16*kHi .. 16*kHi+15 of column n  -> 32 contiguous bytes.
__device__ __forceinline__ v16bf lds_frag_b(const __bf16* p) {
    v8bf a = *(const v8bf*)p;
    v8bf b = *(const v8bf*)(p + 8);
    return __builtin_shufflevector(a, b, 0,1,2,3,4,5,6,7,8,9,10,11,12,13,14,15);
}

#define LSTR 40   // LDS row stride in bf16 elems: 32 + 8 pad (80B, 16B-aligned, conflict-free)

// ---------------------------------------------------------------------------
// C = X * Y  (all 2048x2048 f32 row-major), split-bf16 WMMA.
// Grid: (16,16,nbatch), 256 threads (8 waves). WG tile 128x128.
// Waves 2x4; wave tile 64x32 -> 8 accumulators of 16x16 f32.
// ---------------------------------------------------------------------------
__global__ __launch_bounds__(256)
void powmm_kernel(const float* __restrict__ Xb, const float* __restrict__ Yb,
                  float* __restrict__ Cb,
                  unsigned long long sx, unsigned long long sy, unsigned long long sc)
{
    __shared__ __bf16 AsHi[128 * LSTR];
    __shared__ __bf16 AsLo[128 * LSTR];
    __shared__ __bf16 BtHi[128 * LSTR];
    __shared__ __bf16 BtLo[128 * LSTR];

    const float* X = Xb + (size_t)blockIdx.z * sx;
    const float* Y = Yb + (size_t)blockIdx.z * sy;
    float*       C = Cb + (size_t)blockIdx.z * sc;

    const int tid  = threadIdx.x;
    const int lane = tid & 31;
    const int wave = tid >> 5;
    const int m16  = lane & 15;
    const int kHi  = lane >> 4;

    const int rowBase = blockIdx.y * 128;
    const int colBase = blockIdx.x * 128;
    const int wm = (wave >> 2) * 64;   // wave row offset in tile
    const int wn = (wave & 3) * 32;    // wave col offset in tile

    v8f acc[4][2] = {};

    for (int k0 = 0; k0 < NN2048; k0 += 32) {
        __syncthreads();
        // ---- stage A (128x32) and B (32x128, transposed) as bf16 hi/lo ----
#pragma unroll
        for (int i = 0; i < 4; ++i) {
            const int slot = tid + i * 256;          // 1024 float4 slots each
            // A: 8 float4 per row of 32
            {
                const int r = slot >> 3, c = (slot & 7) * 4;
                const float4 v = *(const float4*)&X[(size_t)(rowBase + r) * NN2048 + k0 + c];
                const float xs[4] = {v.x, v.y, v.z, v.w};
#pragma unroll
                for (int j = 0; j < 4; ++j) {
                    __bf16 hb, lb; split_bf16(xs[j], hb, lb);
                    AsHi[r * LSTR + c + j] = hb;
                    AsLo[r * LSTR + c + j] = lb;
                }
            }
            // B: 32 float4 per row of 128; store transposed
            {
                const int r = slot >> 5, c = (slot & 31) * 4;
                const float4 v = *(const float4*)&Y[(size_t)(k0 + r) * NN2048 + colBase + c];
                const float ys[4] = {v.x, v.y, v.z, v.w};
#pragma unroll
                for (int j = 0; j < 4; ++j) {
                    __bf16 hb, lb; split_bf16(ys[j], hb, lb);
                    BtHi[(c + j) * LSTR + r] = hb;
                    BtLo[(c + j) * LSTR + r] = lb;
                }
            }
        }
        __syncthreads();

        // ---- fragments ----
        v16bf aH[4], aL[4], bH[2], bL[2];
#pragma unroll
        for (int mt = 0; mt < 4; ++mt) {
            const int row = wm + mt * 16 + m16;
            aH[mt] = lds_frag_a(&AsHi[row * LSTR + kHi * 16]);
            aL[mt] = lds_frag_a(&AsLo[row * LSTR + kHi * 16]);
        }
#pragma unroll
        for (int nt = 0; nt < 2; ++nt) {
            const int col = wn + nt * 16 + m16;
            bH[nt] = lds_frag_b(&BtHi[col * LSTR + kHi * 16]);
            bL[nt] = lds_frag_b(&BtLo[col * LSTR + kHi * 16]);
        }

        // ---- 3-term split-precision WMMA: hi*hi + hi*lo + lo*hi ----
#pragma unroll
        for (int mt = 0; mt < 4; ++mt)
#pragma unroll
            for (int nt = 0; nt < 2; ++nt) {
                acc[mt][nt] = __builtin_amdgcn_wmma_f32_16x16x32_bf16(
                    false, aH[mt], false, bH[nt], (short)0, acc[mt][nt], false, false);
                acc[mt][nt] = __builtin_amdgcn_wmma_f32_16x16x32_bf16(
                    false, aH[mt], false, bL[nt], (short)0, acc[mt][nt], false, false);
                acc[mt][nt] = __builtin_amdgcn_wmma_f32_16x16x32_bf16(
                    false, aL[mt], false, bH[nt], (short)0, acc[mt][nt], false, false);
            }
    }

    // ---- epilogue: C/D layout -> M = v + 8*kHi, N = m16 ----
#pragma unroll
    for (int mt = 0; mt < 4; ++mt)
#pragma unroll
        for (int nt = 0; nt < 2; ++nt) {
            const int col = colBase + wn + nt * 16 + m16;
#pragma unroll
            for (int v = 0; v < 8; ++v) {
                const int row = rowBase + wm + mt * 16 + v + 8 * kHi;
                C[(size_t)row * NN2048 + col] = acc[mt][nt][v];
            }
        }
}

// ---------------------------------------------------------------------------
// Per-n fused diagonals + channel combine.
//   d1=A[n,n] d2=rowdot(A,A) d3=rowdot(A,W2) d4=W4[n,n] d5=rowdot(A,W4)
//   d6=W6[n,n] d7=rowdot(A,W6) d8=rowdot(W2,W6)
// out[b,n,c] = h[0,c] + sum_i h[i+1,c]*d[i]
// Grid: (2048,1,nbatch), 256 threads.
// ---------------------------------------------------------------------------
__global__ __launch_bounds__(256)
void diag_out_kernel(const float* __restrict__ Ab,  const float* __restrict__ W2b,
                     const float* __restrict__ W4b, const float* __restrict__ W6b,
                     const float* __restrict__ h,   float* __restrict__ outb,
                     unsigned long long sA, unsigned long long sW)
{
    const float* A  = Ab  + (size_t)blockIdx.z * sA;
    const float* W2 = W2b + (size_t)blockIdx.z * sW;
    const float* W4 = W4b + (size_t)blockIdx.z * sW;
    const float* W6 = W6b + (size_t)blockIdx.z * sW;
    float*      out = outb + (size_t)blockIdx.z * (size_t)NN2048 * NCH;

    const int n = blockIdx.x, tid = threadIdx.x;
    float s2 = 0.f, s3 = 0.f, s5 = 0.f, s7 = 0.f, s8 = 0.f;
    for (int k = tid; k < NN2048; k += 256) {
        const float a_nk  = A [(size_t)n * NN2048 + k];
        const float w2_nk = W2[(size_t)n * NN2048 + k];
        const float a_kn  = A [(size_t)k * NN2048 + n];
        const float w2_kn = W2[(size_t)k * NN2048 + n];
        const float w4_kn = W4[(size_t)k * NN2048 + n];
        const float w6_kn = W6[(size_t)k * NN2048 + n];
        s2 = fmaf(a_nk,  a_kn,  s2);
        s3 = fmaf(a_nk,  w2_kn, s3);
        s5 = fmaf(a_nk,  w4_kn, s5);
        s7 = fmaf(a_nk,  w6_kn, s7);
        s8 = fmaf(w2_nk, w6_kn, s8);
    }
#pragma unroll
    for (int off = 16; off > 0; off >>= 1) {   // wave32 butterfly
        s2 += __shfl_xor(s2, off);
        s3 += __shfl_xor(s3, off);
        s5 += __shfl_xor(s5, off);
        s7 += __shfl_xor(s7, off);
        s8 += __shfl_xor(s8, off);
    }
    __shared__ float part[8][5];
    __shared__ float d[8];
    const int lane = tid & 31, wave = tid >> 5;
    if (lane == 0) {
        part[wave][0] = s2; part[wave][1] = s3; part[wave][2] = s5;
        part[wave][3] = s7; part[wave][4] = s8;
    }
    __syncthreads();
    if (tid < 5) {
        float s = 0.f;
#pragma unroll
        for (int w = 0; w < 8; ++w) s += part[w][tid];
        const int map[5] = {1, 2, 4, 6, 7};    // -> d2,d3,d5,d7,d8
        d[map[tid]] = s;
    }
    if (tid == 5) d[0] = A [(size_t)n * (NN2048 + 1)];
    if (tid == 6) d[3] = W4[(size_t)n * (NN2048 + 1)];
    if (tid == 7) d[5] = W6[(size_t)n * (NN2048 + 1)];
    __syncthreads();
    if (tid < NCH) {
        const int c = tid;
        float v = h[c];                         // h[0,c]
#pragma unroll
        for (int i = 0; i < 8; ++i) v = fmaf(h[(i + 1) * NCH + c], d[i], v);
        out[(size_t)n * NCH + c] = v;
    }
}

// ---------------------------------------------------------------------------
extern "C" void kernel_launch(void* const* d_in, const int* in_sizes, int n_in,
                              void* d_out, int out_size, void* d_ws, size_t ws_size,
                              hipStream_t stream) {
    const float* A = (const float*)d_in[0];   // [8,2048,2048]
    const float* h = (const float*)d_in[1];   // [9,64]
    float* out = (float*)d_out;               // [8,2048,64]

    const size_t NNq  = (size_t)NN2048 * NN2048;
    const size_t full = 3ull * BATCH * NNq * sizeof(float);
    const dim3 blk(256);

    if (ws_size >= full) {
        // all batches in flight: W2 | W4 | W6, each [8, N, N]
        float* W2 = (float*)d_ws;
        float* W4 = W2 + BATCH * NNq;
        float* W6 = W4 + BATCH * NNq;
        const dim3 g(NN2048 / 128, NN2048 / 128, BATCH);
        powmm_kernel<<<g, blk, 0, stream>>>(A,  A,  W2, NNq, NNq, NNq);
        powmm_kernel<<<g, blk, 0, stream>>>(W2, W2, W4, NNq, NNq, NNq);
        powmm_kernel<<<g, blk, 0, stream>>>(W2, W4, W6, NNq, NNq, NNq);
        diag_out_kernel<<<dim3(NN2048, 1, BATCH), blk, 0, stream>>>(
            A, W2, W4, W6, h, out, NNq, NNq);
    } else {
        // per-batch workspace (3 * 16.8 MB)
        float* W2 = (float*)d_ws;
        float* W4 = W2 + NNq;
        float* W6 = W4 + NNq;
        const dim3 g(NN2048 / 128, NN2048 / 128, 1);
        for (int b = 0; b < BATCH; ++b) {
            const float* Abp = A + (size_t)b * NNq;
            powmm_kernel<<<g, blk, 0, stream>>>(Abp, Abp, W2, 0, 0, 0);
            powmm_kernel<<<g, blk, 0, stream>>>(W2,  W2,  W4, 0, 0, 0);
            powmm_kernel<<<g, blk, 0, stream>>>(W2,  W4,  W6, 0, 0, 0);
            diag_out_kernel<<<dim3(NN2048, 1, 1), blk, 0, stream>>>(
                Abp, W2, W4, W6, h, out + (size_t)b * NN2048 * NCH, 0, 0);
        }
    }
}
